// L4Block_24850680775020
// MI455X (gfx1250) — matrix operand order repaired and checked by reading, not compiled, not run on hardware
//
#include <hip/hip_runtime.h>

// ---------------------------------------------------------------------------
// Llama-4-style block for MI455X (gfx1250, wave32, WMMA 16x16x32 bf16).
//  - Weights pre-converted f32->bf16 once so gate+up (67MB bf16) stay L2-resident
//    across the 32 M-tile sweeps (192MB global L2).
//  - GEMM B-tiles and attention K/V^T tiles streamed into LDS by the Tensor
//    Data Mover (tensor_load_to_lds + s_wait_tensorcnt), double-buffered.
//  - Attention: 64-row query block per workgroup (4 waves) sharing each K/V
//    tile from LDS -> 4x less L2 traffic; flash-style online softmax; all WMMA.
// ---------------------------------------------------------------------------

typedef __attribute__((ext_vector_type(16))) __bf16 v16bf;
typedef __attribute__((ext_vector_type(8)))  __bf16 v8bf;
typedef __attribute__((ext_vector_type(8)))  float  v8f;
typedef __attribute__((ext_vector_type(4)))  unsigned int v4u;
typedef __attribute__((ext_vector_type(4)))  int  v4i;
typedef __attribute__((ext_vector_type(8)))  int  v8i;

union BF16x16 { v16bf v; v8bf h[2]; };

static constexpr int T_SEQ  = 4096;
static constexpr int C_DIM  = 2048;
static constexpr int N_H    = 16;
static constexpr int N_KVH  = 4;
static constexpr int HD     = 128;
static constexpr int KV_DIM = N_KVH * HD;   // 512
static constexpr int FFN_H  = 8192;

#if defined(__has_builtin)
#  if __has_builtin(__builtin_amdgcn_tensor_load_to_lds)
#    define USE_TDM 1
#  endif
#endif
#ifndef USE_TDM
#  define USE_TDM 0
#endif

__device__ __forceinline__ v8f wmma_bf16(const v16bf& a, const v16bf& b, const v8f& c) {
    return __builtin_amdgcn_wmma_f32_16x16x32_bf16(false, a, false, b, (short)0, c, false, false);
}

#if USE_TDM
// TDM 2-D bf16 tile load: tile_h rows x tile_w elems, row stride `stride` elems.
// D# packing per CDNA5 ISA (group0 128b, group1 256b).
__device__ __forceinline__ void tdm_load_tile(const __bf16* gsrc, void* lds_dst,
                                              int tile_w, int tile_h, int stride) {
    const unsigned long long ga = (unsigned long long)(uintptr_t)gsrc;
    const unsigned int lds = (unsigned int)(uintptr_t)lds_dst;   // flat low 32 = LDS offset
    v4u g0 = { 1u,                                        // count=1, user descriptor
               lds,                                       // lds_addr
               (unsigned int)ga,                          // global_addr[31:0]
               (unsigned int)(ga >> 32) | 0x80000000u };  // addr[56:32] | type=2
    v8i g1 = { (int)(1u << 16),                                       // data_size=1 (2B)
               (int)(((unsigned)stride & 0xffffu) << 16),             // tensor_dim0 lo16
               (int)((((unsigned)stride >> 16) & 0xffffu) | 0x20000000u), // dim0 hi | dim1 lo (8192)
               (int)((unsigned)tile_w << 16),                         // dim1 hi=0 | tile_dim0
               tile_h,                                                // tile_dim1 | tile_dim2=0
               stride,                                                // tensor_dim0_stride lo32
               0, 0 };                                                // stride hi / dim1_stride
    v4i z4 = { 0, 0, 0, 0 };
#if __clang_major__ >= 23
    v8i z8 = { 0, 0, 0, 0, 0, 0, 0, 0 };
    __builtin_amdgcn_tensor_load_to_lds(g0, g1, z4, z4, z8, 0);
#else
    __builtin_amdgcn_tensor_load_to_lds(g0, g1, z4, z4, 0);
#endif
}
#endif

// ---------------------------------------------------------------------------
// f32 -> bf16 weight conversion (one pass; keeps weights L2-resident as bf16)
// ---------------------------------------------------------------------------
__global__ __launch_bounds__(256) void cvt_bf16_kernel(const float* __restrict__ src,
                                                       __bf16* __restrict__ dst, int n) {
    const int i = (blockIdx.x * 256 + threadIdx.x) * 4;
    if (i >= n) return;
    const float4 f = *(const float4*)(src + i);
    dst[i + 0] = (__bf16)f.x; dst[i + 1] = (__bf16)f.y;
    dst[i + 2] = (__bf16)f.z; dst[i + 3] = (__bf16)f.w;
}

// ---------------------------------------------------------------------------
// RMSNorm: x[row][C] (f32) -> out[row][C] (bf16).  One block per row.
// ---------------------------------------------------------------------------
__global__ __launch_bounds__(256) void rmsnorm_kernel(const float* __restrict__ x,
                                                      const float* __restrict__ w,
                                                      __bf16* __restrict__ out) {
    const int row = blockIdx.x;
    const int tid = threadIdx.x;
    const float* xr = x + (size_t)row * C_DIM;

    float4 v0 = ((const float4*)xr)[tid * 2 + 0];
    float4 v1 = ((const float4*)xr)[tid * 2 + 1];
    float s = v0.x*v0.x + v0.y*v0.y + v0.z*v0.z + v0.w*v0.w
            + v1.x*v1.x + v1.y*v1.y + v1.z*v1.z + v1.w*v1.w;
    #pragma unroll
    for (int m = 16; m >= 1; m >>= 1) s += __shfl_xor(s, m, 32);

    __shared__ float part[8];
    if ((tid & 31) == 0) part[tid >> 5] = s;
    __syncthreads();
    float tot = 0.f;
    #pragma unroll
    for (int i = 0; i < 8; ++i) tot += part[i];
    const float sc = rsqrtf(tot * (1.0f / (float)C_DIM) + 1e-5f);

    float4 w0 = ((const float4*)w)[tid * 2 + 0];
    float4 w1 = ((const float4*)w)[tid * 2 + 1];
    __bf16* o = out + (size_t)row * C_DIM + tid * 8;
    o[0] = (__bf16)(v0.x * sc * w0.x);  o[1] = (__bf16)(v0.y * sc * w0.y);
    o[2] = (__bf16)(v0.z * sc * w0.z);  o[3] = (__bf16)(v0.w * sc * w0.w);
    o[4] = (__bf16)(v1.x * sc * w1.x);  o[5] = (__bf16)(v1.y * sc * w1.y);
    o[6] = (__bf16)(v1.z * sc * w1.z);  o[7] = (__bf16)(v1.w * sc * w1.w);
}

// ---------------------------------------------------------------------------
// GEMM: out[M][N] = A(bf16)[M][K] @ Wb(bf16,[N][K])^T  (+resid) (+SwiGLU)
// Block = 128 threads (4 waves). Tile = 128(M) x 64(N), K-step 32.
// Wave owns 32 M rows: 2 A-frags x 4 B-frags = 8 WMMA per K-step.
// B tile fetched by TDM into double-buffered LDS (s_wait_tensorcnt),
// cooperative-copy fallback if the builtin is unavailable.
// ---------------------------------------------------------------------------
template <bool GATEUP, bool RESID, bool OUTBF>
__global__ __launch_bounds__(128) void gemm_bf16_kernel(const __bf16* __restrict__ A,
                                                        const __bf16* __restrict__ Wb,
                                                        const __bf16* __restrict__ W2b,
                                                        const float*  __restrict__ resid,
                                                        void* __restrict__ outp,
                                                        int M, int N, int K) {
    constexpr int TN = 64, TK = 32;
    constexpr int NBUF = USE_TDM ? 2 : 1;
    __shared__ __bf16 bshW[NBUF * TN * TK];
    __shared__ __bf16 bshU[GATEUP ? (NBUF * TN * TK) : 1];

    const int tid  = threadIdx.x;
    const int wid  = tid >> 5;
    const int lane = tid & 31;
    const int lo   = lane & 15, hi = lane >> 4;
    const int nb   = blockIdx.x * TN;
    const int mb   = blockIdx.y * 128;
    const int wrow = mb + wid * 32;

    v8f acc[2][4];
    v8f accU[GATEUP ? 2 : 1][4];
    #pragma unroll
    for (int h = 0; h < 2; ++h)
        #pragma unroll
        for (int c = 0; c < 4; ++c) acc[h][c] = (v8f)0.f;
    if constexpr (GATEUP) {
        #pragma unroll
        for (int h = 0; h < 2; ++h)
            #pragma unroll
            for (int c = 0; c < 4; ++c) accU[h][c] = (v8f)0.f;
    }

    const int nk = K / TK;

#if USE_TDM
    if (wid == 0) {
        tdm_load_tile(Wb + (size_t)nb * K, &bshW[0], TK, TN, K);
        if constexpr (GATEUP) tdm_load_tile(W2b + (size_t)nb * K, &bshU[0], TK, TN, K);
    }
#endif

    for (int i = 0; i < nk; ++i) {
        const int kb = i * TK;
#if USE_TDM
        const int cur = i & 1;
        if (wid == 0) {
            if (i + 1 < nk) {
                tdm_load_tile(Wb + (size_t)nb * K + kb + TK, &bshW[(cur ^ 1) * TN * TK], TK, TN, K);
                if constexpr (GATEUP)
                    tdm_load_tile(W2b + (size_t)nb * K + kb + TK, &bshU[(cur ^ 1) * TN * TK], TK, TN, K);
                __builtin_amdgcn_s_wait_tensorcnt(GATEUP ? 2 : 1);
            } else {
                __builtin_amdgcn_s_wait_tensorcnt(0);
            }
        }
        __syncthreads();
#else
        const int cur = 0;
        __syncthreads();
        {   // cooperative bf16 tile copy: thread -> (row r, K-half)
            const int r = tid >> 1, half = tid & 1;
            const __bf16* src = Wb + (size_t)(nb + r) * K + kb + half * 16;
            *(v8bf*)&bshW[r * TK + half * 16 + 0] = *(const v8bf*)(src + 0);
            *(v8bf*)&bshW[r * TK + half * 16 + 8] = *(const v8bf*)(src + 8);
            if (kb + TK < K) __builtin_prefetch(src + TK, 0, 0);
            if constexpr (GATEUP) {
                const __bf16* s2 = W2b + (size_t)(nb + r) * K + kb + half * 16;
                *(v8bf*)&bshU[r * TK + half * 16 + 0] = *(const v8bf*)(s2 + 0);
                *(v8bf*)&bshU[r * TK + half * 16 + 8] = *(const v8bf*)(s2 + 8);
                if (kb + TK < K) __builtin_prefetch(s2 + TK, 0, 0);
            }
        }
        __syncthreads();
#endif
        const __bf16* arow0 = A + (size_t)(wrow + lo) * K + kb + hi * 8;
        const __bf16* arow1 = arow0 + (size_t)16 * K;
        BF16x16 a0, a1;
        a0.h[0] = *(const v8bf*)(arow0);  a0.h[1] = *(const v8bf*)(arow0 + 16);
        a1.h[0] = *(const v8bf*)(arow1);  a1.h[1] = *(const v8bf*)(arow1 + 16);

        const __bf16* bufW = &bshW[cur * TN * TK];
        #pragma unroll
        for (int c = 0; c < 4; ++c) {
            const __bf16* bptr = bufW + (c * 16 + lo) * TK + hi * 8;
            BF16x16 b;
            b.h[0] = *(const v8bf*)(bptr);
            b.h[1] = *(const v8bf*)(bptr + 16);
            acc[0][c] = wmma_bf16(a0.v, b.v, acc[0][c]);
            acc[1][c] = wmma_bf16(a1.v, b.v, acc[1][c]);
            if constexpr (GATEUP) {
                const __bf16* b2p = &bshU[cur * TN * TK] + (c * 16 + lo) * TK + hi * 8;
                BF16x16 b2;
                b2.h[0] = *(const v8bf*)(b2p);
                b2.h[1] = *(const v8bf*)(b2p + 16);
                accU[0][c] = wmma_bf16(a0.v, b2.v, accU[0][c]);
                accU[1][c] = wmma_bf16(a1.v, b2.v, accU[1][c]);
            }
        }
#if USE_TDM
        __syncthreads();   // reads done before this buffer is re-targeted by TDM
#endif
    }

    #pragma unroll
    for (int h = 0; h < 2; ++h) {
        #pragma unroll
        for (int c = 0; c < 4; ++c) {
            #pragma unroll
            for (int j = 0; j < 8; ++j) {
                const int row = wrow + h * 16 + j + 8 * hi;
                const int col = nb + c * 16 + lo;
                float val = acc[h][c][j];
                if constexpr (RESID) val += resid[(size_t)row * N + col];
                if constexpr (GATEUP) {
                    const float g = val, u = accU[h][c][j];
                    const float sig = 1.0f / (1.0f + __expf(-g));
                    ((__bf16*)outp)[(size_t)row * N + col] = (__bf16)(g * sig * u);
                } else if constexpr (OUTBF) {
                    ((__bf16*)outp)[(size_t)row * N + col] = (__bf16)val;
                } else {
                    ((float*)outp)[(size_t)row * N + col] = val;
                }
            }
        }
    }
}

// ---------------------------------------------------------------------------
// RoPE (interleaved pairs) applied in place to q[T][2048] and k[T][512] (bf16).
// ---------------------------------------------------------------------------
__global__ __launch_bounds__(256) void rope_kernel(__bf16* __restrict__ qr,
                                                   __bf16* __restrict__ kr) {
    const int id = blockIdx.x * 256 + threadIdx.x;
    const int total = T_SEQ * (N_H + N_KVH) * (HD / 2);
    if (id >= total) return;
    const int t = id / 1280;
    const int p = id % 1280;
    __bf16* ptr;
    int i;
    if (p < N_H * 64) {
        const int head = p >> 6; i = p & 63;
        ptr = qr + (size_t)t * C_DIM + head * HD + 2 * i;
    } else {
        const int kvh = (p - N_H * 64) >> 6; i = p & 63;
        ptr = kr + (size_t)t * KV_DIM + kvh * HD + 2 * i;
    }
    const float inv_freq = __expf(-((float)(2 * i) * (1.0f / 128.0f)) * __logf(500000.0f));
    const float ang = (float)t * inv_freq;
    float c, s;
    __sincosf(ang, &s, &c);
    const float a = (float)ptr[0], b = (float)ptr[1];
    ptr[0] = (__bf16)(a * c - b * s);
    ptr[1] = (__bf16)(a * s + b * c);
}

// ---------------------------------------------------------------------------
// V transpose: vr[T][512] -> vt[512][T] (bf16)
// ---------------------------------------------------------------------------
__global__ __launch_bounds__(256) void transpose_v_kernel(const __bf16* __restrict__ vr,
                                                          __bf16* __restrict__ vt) {
    const int id = blockIdx.x * 256 + threadIdx.x;
    if (id >= KV_DIM * T_SEQ) return;
    const int c = id >> 12;
    const int t = id & (T_SEQ - 1);
    vt[id] = vr[(size_t)t * KV_DIM + c];
}

// ---------------------------------------------------------------------------
// Flash attention: 64-row query block per workgroup (4 waves x 16 rows), causal,
// GQA (kvh = head/4). Each 32-key K tile (32x128) and V^T tile (128x32) is
// staged ONCE into LDS (TDM double-buffered) and shared by all 4 waves ->
// 4x less L2 traffic. QK^T and PV via bf16 WMMA; P relayout through LDS
// (per-wave LDS ops are in-order, no barrier needed for it).
// ---------------------------------------------------------------------------
__global__ __launch_bounds__(128) void attn_kernel(const __bf16* __restrict__ qr,
                                                   const __bf16* __restrict__ kr,
                                                   const __bf16* __restrict__ vt,
                                                   __bf16* __restrict__ y) {
    constexpr int NBUF = USE_TDM ? 2 : 1;
    __shared__ __bf16 ksh[NBUF][32 * 128];   // [key][dim]
    __shared__ __bf16 vsh[NBUF][128 * 32];   // [dim][key]
    __shared__ __bf16 pbuf[4][16 * 32];

    const int tid  = threadIdx.x;
    const int wid  = tid >> 5;
    const int lane = tid & 31;
    const int lo = lane & 15, hi = lane >> 4;
    const int head = blockIdx.y;
    const int kvh  = head >> 2;
    const int q0   = blockIdx.x * 64;        // block query base
    const int wq0  = q0 + wid * 16;          // wave query base
    const int wmax = wq0 + 15;
    const float scale = 0.08838834764831845f;   // 1/sqrt(128)

    BF16x16 qf[4];
    #pragma unroll
    for (int ks = 0; ks < 4; ++ks) {
        const __bf16* p = qr + (size_t)(wq0 + lo) * C_DIM + head * HD + ks * 32 + hi * 8;
        qf[ks].h[0] = *(const v8bf*)(p);
        qf[ks].h[1] = *(const v8bf*)(p + 16);
    }

    v8f o[8];
    #pragma unroll
    for (int c = 0; c < 8; ++c) o[c] = (v8f)0.f;
    float mst[8], lst[8];
    #pragma unroll
    for (int j = 0; j < 8; ++j) { mst[j] = -1e30f; lst[j] = 0.f; }

    const int nkt = (q0 + 64 + 31) >> 5;     // uniform across the block

#if USE_TDM
    if (wid == 0) {
        tdm_load_tile(kr + 0 * KV_DIM + kvh * HD, &ksh[0][0], HD, 32, KV_DIM);
        tdm_load_tile(vt + (size_t)(kvh * HD) * T_SEQ + 0, &vsh[0][0], 32, HD, T_SEQ);
    }
#endif

    for (int kt = 0; kt < nkt; ++kt) {
        const int kb = kt * 32;
#if USE_TDM
        const int cur = kt & 1;
        if (wid == 0) {
            if (kt + 1 < nkt) {
                tdm_load_tile(kr + (size_t)(kb + 32) * KV_DIM + kvh * HD,
                              &ksh[cur ^ 1][0], HD, 32, KV_DIM);
                tdm_load_tile(vt + (size_t)(kvh * HD) * T_SEQ + kb + 32,
                              &vsh[cur ^ 1][0], 32, HD, T_SEQ);
                __builtin_amdgcn_s_wait_tensorcnt(2);
            } else {
                __builtin_amdgcn_s_wait_tensorcnt(0);
            }
        }
        __syncthreads();
#else
        const int cur = 0;
        __syncthreads();
        {   // cooperative staging: K tile (32 keys x 128 dims), V^T tile (128 x 32)
            const int kr_row = tid >> 2, kq = tid & 3;   // key row, 32-elem quarter
            const __bf16* ks_src = kr + (size_t)(kb + kr_row) * KV_DIM + kvh * HD + kq * 32;
            *(v8bf*)&ksh[0][kr_row * 128 + kq * 32 + 0]  = *(const v8bf*)(ks_src + 0);
            *(v8bf*)&ksh[0][kr_row * 128 + kq * 32 + 8]  = *(const v8bf*)(ks_src + 8);
            *(v8bf*)&ksh[0][kr_row * 128 + kq * 32 + 16] = *(const v8bf*)(ks_src + 16);
            *(v8bf*)&ksh[0][kr_row * 128 + kq * 32 + 24] = *(const v8bf*)(ks_src + 24);
            const __bf16* vs_src = vt + (size_t)(kvh * HD + tid) * T_SEQ + kb;
            *(v8bf*)&vsh[0][tid * 32 + 0]  = *(const v8bf*)(vs_src + 0);
            *(v8bf*)&vsh[0][tid * 32 + 8]  = *(const v8bf*)(vs_src + 8);
            *(v8bf*)&vsh[0][tid * 32 + 16] = *(const v8bf*)(vs_src + 16);
            *(v8bf*)&vsh[0][tid * 32 + 24] = *(const v8bf*)(vs_src + 24);
        }
        __syncthreads();
#endif
        if (kb <= wmax) {   // wave-uniform: skip tiles fully above this wave's rows
            v8f s0 = (v8f)0.f, s1 = (v8f)0.f;
            #pragma unroll
            for (int ks = 0; ks < 4; ++ks) {
                const __bf16* b0p = &ksh[cur][(lo)      * 128 + ks * 32 + hi * 8];
                const __bf16* b1p = &ksh[cur][(16 + lo) * 128 + ks * 32 + hi * 8];
                BF16x16 b0, b1;
                b0.h[0] = *(const v8bf*)(b0p); b0.h[1] = *(const v8bf*)(b0p + 16);
                b1.h[0] = *(const v8bf*)(b1p); b1.h[1] = *(const v8bf*)(b1p + 16);
                s0 = wmma_bf16(qf[ks].v, b0.v, s0);
                s1 = wmma_bf16(qf[ks].v, b1.v, s1);
            }

            float p0[8], p1[8], alpha[8];
            #pragma unroll
            for (int j = 0; j < 8; ++j) {
                const int grow = wq0 + j + 8 * hi;
                float a0 = s0[j] * scale; if (kb + lo      > grow) a0 = -1e30f;
                float a1 = s1[j] * scale; if (kb + 16 + lo > grow) a1 = -1e30f;
                float t = fmaxf(a0, a1);
                t = fmaxf(t, __shfl_xor(t, 1, 32));
                t = fmaxf(t, __shfl_xor(t, 2, 32));
                t = fmaxf(t, __shfl_xor(t, 4, 32));
                t = fmaxf(t, __shfl_xor(t, 8, 32));
                const float newm = fmaxf(mst[j], t);
                alpha[j] = __expf(mst[j] - newm);
                p0[j] = __expf(a0 - newm);
                p1[j] = __expf(a1 - newm);
                float ps = p0[j] + p1[j];
                ps += __shfl_xor(ps, 1, 32);
                ps += __shfl_xor(ps, 2, 32);
                ps += __shfl_xor(ps, 4, 32);
                ps += __shfl_xor(ps, 8, 32);
                lst[j] = lst[j] * alpha[j] + ps;
                mst[j] = newm;
            }
            #pragma unroll
            for (int c = 0; c < 8; ++c)
                #pragma unroll
                for (int j = 0; j < 8; ++j) o[c][j] *= alpha[j];

            #pragma unroll
            for (int j = 0; j < 8; ++j) {
                pbuf[wid][(j + 8 * hi) * 32 + lo]      = (__bf16)p0[j];
                pbuf[wid][(j + 8 * hi) * 32 + 16 + lo] = (__bf16)p1[j];
            }
            BF16x16 pf;
            {
                const __bf16* pp = &pbuf[wid][lo * 32 + hi * 8];
                pf.h[0] = *(const v8bf*)(pp);
                pf.h[1] = *(const v8bf*)(pp + 16);
            }

            #pragma unroll
            for (int c = 0; c < 8; ++c) {
                const __bf16* vp = &vsh[cur][(c * 16 + lo) * 32 + hi * 8];
                BF16x16 b;
                b.h[0] = *(const v8bf*)(vp);
                b.h[1] = *(const v8bf*)(vp + 16);
                o[c] = wmma_bf16(pf.v, b.v, o[c]);
            }
        }
        __syncthreads();   // tile reads done before buffer is re-targeted
    }

    #pragma unroll
    for (int c = 0; c < 8; ++c) {
        #pragma unroll
        for (int j = 0; j < 8; ++j) {
            const int row = wq0 + j + 8 * hi;
            const int col = head * HD + c * 16 + lo;
            y[(size_t)row * C_DIM + col] = (__bf16)(o[c][j] * (1.0f / lst[j]));
        }
    }
}

// ---------------------------------------------------------------------------
// Host-side orchestration
// ---------------------------------------------------------------------------
extern "C" void kernel_launch(void* const* d_in, const int* in_sizes, int n_in,
                              void* d_out, int out_size, void* d_ws, size_t ws_size,
                              hipStream_t stream) {
    (void)in_sizes; (void)n_in; (void)out_size; (void)ws_size;

    const float* x        = (const float*)d_in[0];
    // d_in[1] = attn_bias: pure causal mask, applied analytically in attn_kernel
    const float* attn_nw  = (const float*)d_in[2];
    const float* wq       = (const float*)d_in[3];
    const float* wk       = (const float*)d_in[4];
    const float* wv       = (const float*)d_in[5];
    const float* wo       = (const float*)d_in[6];
    const float* ffn_nw   = (const float*)d_in[7];
    const float* w_gate   = (const float*)d_in[8];
    const float* w_up     = (const float*)d_in[9];
    const float* w_down   = (const float*)d_in[10];
    float* out = (float*)d_out;

    char* wsb = (char*)d_ws;
    auto carve = [&wsb](size_t bytes) -> void* {
        void* p = (void*)wsb;
        wsb += (bytes + 255) & ~(size_t)255;
        return p;
    };
    __bf16* h_bf   = (__bf16*)carve((size_t)T_SEQ * C_DIM  * 2);
    __bf16* qr     = (__bf16*)carve((size_t)T_SEQ * C_DIM  * 2);
    __bf16* kr     = (__bf16*)carve((size_t)T_SEQ * KV_DIM * 2);
    __bf16* vr     = (__bf16*)carve((size_t)T_SEQ * KV_DIM * 2);
    __bf16* vt     = (__bf16*)carve((size_t)KV_DIM * T_SEQ * 2);
    __bf16* y_bf   = (__bf16*)carve((size_t)T_SEQ * C_DIM  * 2);
    float*  x1     = (float* )carve((size_t)T_SEQ * C_DIM  * 4);
    __bf16* h2_bf  = (__bf16*)carve((size_t)T_SEQ * C_DIM  * 2);
    __bf16* act    = (__bf16*)carve((size_t)T_SEQ * FFN_H  * 2);
    __bf16* wq_b   = (__bf16*)carve((size_t)C_DIM  * C_DIM * 2);
    __bf16* wk_b   = (__bf16*)carve((size_t)KV_DIM * C_DIM * 2);
    __bf16* wv_b   = (__bf16*)carve((size_t)KV_DIM * C_DIM * 2);
    __bf16* wo_b   = (__bf16*)carve((size_t)C_DIM  * C_DIM * 2);
    __bf16* wg_b   = (__bf16*)carve((size_t)FFN_H  * C_DIM * 2);
    __bf16* wu_b   = (__bf16*)carve((size_t)FFN_H  * C_DIM * 2);
    __bf16* wd_b   = (__bf16*)carve((size_t)C_DIM  * FFN_H * 2);

    auto cvt = [&](const float* s, __bf16* d, int n) {
        cvt_bf16_kernel<<<dim3(n / 4 / 256), dim3(256), 0, stream>>>(s, d, n);
    };
    cvt(wq,     wq_b, C_DIM * C_DIM);
    cvt(wk,     wk_b, KV_DIM * C_DIM);
    cvt(wv,     wv_b, KV_DIM * C_DIM);
    cvt(wo,     wo_b, C_DIM * C_DIM);
    cvt(w_gate, wg_b, FFN_H * C_DIM);
    cvt(w_up,   wu_b, FFN_H * C_DIM);
    cvt(w_down, wd_b, C_DIM * FFN_H);

    rmsnorm_kernel<<<dim3(T_SEQ), dim3(256), 0, stream>>>(x, attn_nw, h_bf);

    gemm_bf16_kernel<false, false, true><<<dim3(C_DIM / 64, T_SEQ / 128), dim3(128), 0, stream>>>(
        h_bf, wq_b, nullptr, nullptr, (void*)qr, T_SEQ, C_DIM, C_DIM);
    gemm_bf16_kernel<false, false, true><<<dim3(KV_DIM / 64, T_SEQ / 128), dim3(128), 0, stream>>>(
        h_bf, wk_b, nullptr, nullptr, (void*)kr, T_SEQ, KV_DIM, C_DIM);
    gemm_bf16_kernel<false, false, true><<<dim3(KV_DIM / 64, T_SEQ / 128), dim3(128), 0, stream>>>(
        h_bf, wv_b, nullptr, nullptr, (void*)vr, T_SEQ, KV_DIM, C_DIM);

    {
        const int total = T_SEQ * (N_H + N_KVH) * (HD / 2);
        rope_kernel<<<dim3((total + 255) / 256), dim3(256), 0, stream>>>(qr, kr);
    }
    transpose_v_kernel<<<dim3((KV_DIM * T_SEQ + 255) / 256), dim3(256), 0, stream>>>(vr, vt);

    attn_kernel<<<dim3(T_SEQ / 64, N_H), dim3(128), 0, stream>>>(qr, kr, vt, y_bf);

    gemm_bf16_kernel<false, true, false><<<dim3(C_DIM / 64, T_SEQ / 128), dim3(128), 0, stream>>>(
        y_bf, wo_b, nullptr, x, (void*)x1, T_SEQ, C_DIM, C_DIM);

    rmsnorm_kernel<<<dim3(T_SEQ), dim3(256), 0, stream>>>(x1, ffn_nw, h2_bf);

    gemm_bf16_kernel<true, false, true><<<dim3(FFN_H / 64, T_SEQ / 128), dim3(128), 0, stream>>>(
        h2_bf, wg_b, wu_b, nullptr, (void*)act, T_SEQ, FFN_H, C_DIM);

    gemm_bf16_kernel<false, true, false><<<dim3(C_DIM / 64, T_SEQ / 128), dim3(128), 0, stream>>>(
        act, wd_b, nullptr, x1, (void*)out, T_SEQ, C_DIM, FFN_H);
}
